// MarkovwthStopembeddingwithLSTM_27410481283756
// MI455X (gfx1250) — matrix operand order repaired and checked by reading, not compiled, not run on hardware
//
#include <hip/hip_runtime.h>
#include <hip/hip_bf16.h>

// ---------------------------------------------------------------------------
// Types for WMMA
// ---------------------------------------------------------------------------
typedef __attribute__((ext_vector_type(16))) __bf16 v16bf;
typedef __attribute__((ext_vector_type(8)))  float  v8f;

union FragAB {
    v16bf v;
    uint4 q[2];
};

// ---------------------------------------------------------------------------
// Problem constants
// ---------------------------------------------------------------------------
#define BATCH 1024
#define TSTEPS 60
#define NNODES 512       // LSTM hidden
#define KTOT 1024        // fused K = 512 (x) + 512 (h)
#define JTOT 2048        // 4 gates * 512 nodes, j' = 64*(n>>4)+16*g+(n&15)
#define EDIM 64
#define FDIM 8
#define SEDIM 64
#define SMAX 40

// Workspace layout (bytes, all 256-aligned)
#define XBF_OFF   0ULL                                    // [B,T,512] bf16
#define WCAT_OFF  (XBF_OFF + 62914560ULL)                 // [2048,1024] bf16 (B^T)
#define BIAS_OFF  (WCAT_OFF + 4194304ULL)                 // [2048] f32
#define HA_OFF    (BIAS_OFF + 8192ULL)                    // [B,512] bf16
#define HB_OFF    (HA_OFF + 1048576ULL)                   // [B,512] bf16
#define C_OFF     (HB_OFF + 1048576ULL)                   // [B,512] f32
#define HF_OFF    (C_OFF + 2097152ULL)                    // [B,512] f32

// ---------------------------------------------------------------------------
// CDNA5 async global->LDS copy (ASYNCcnt-tracked), 16 bytes per lane, with
// the byte offset folded into the instruction's 24-bit IOFFSET field so no
// per-issue address VALU is needed. No memory clobber: ordering vs consumers
// is enforced by the volatile s_wait_asynccnt + __syncthreads() fences.
// ---------------------------------------------------------------------------
__device__ __forceinline__ void async_b128_off(unsigned lds_off, const void* gbase,
                                               int ioff) {
    asm volatile("global_load_async_to_lds_b128 %0, %1, off offset:%c2"
                 :: "v"(lds_off), "v"(gbase), "i"(ioff));
}

// ---------------------------------------------------------------------------
// Prep: x fp32 -> bf16 (4 elems/thread)
// ---------------------------------------------------------------------------
__global__ __launch_bounds__(256) void prep_x_kernel(const float* __restrict__ x,
                                                     __hip_bfloat16* __restrict__ xbf,
                                                     int n4) {
    int i = blockIdx.x * 256 + threadIdx.x;
    if (i < n4) {
        float4 v = reinterpret_cast<const float4*>(x)[i];
        xbf[4 * i + 0] = __float2bfloat16(v.x);
        xbf[4 * i + 1] = __float2bfloat16(v.y);
        xbf[4 * i + 2] = __float2bfloat16(v.z);
        xbf[4 * i + 3] = __float2bfloat16(v.w);
    }
}

// ---------------------------------------------------------------------------
// Prep: fused, gate-blocked, transposed weights in bf16.
// Column j' = 64*(n>>4) + 16*g + (n&15): the 4 gates of node n land in the
// same lane of 4 consecutive 16-wide N-fragments -> register-only epilogue.
// ---------------------------------------------------------------------------
__global__ __launch_bounds__(256) void prep_w_kernel(const float* __restrict__ W_ih,
                                                     const float* __restrict__ W_hh,
                                                     __hip_bfloat16* __restrict__ wcatT) {
    int idx = blockIdx.x * 256 + threadIdx.x;       // < 2048*1024
    int j = idx >> 10;
    int k = idx & 1023;
    int q = j >> 6;
    int rem = j & 63;
    int g = rem >> 4;
    int m = rem & 15;
    int n = q * 16 + m;
    int row = g * NNODES + n;
    float v = (k < NNODES) ? W_ih[(size_t)row * NNODES + k]
                           : W_hh[(size_t)row * NNODES + (k - NNODES)];
    wcatT[(size_t)j * KTOT + k] = __float2bfloat16(v);
}

__global__ __launch_bounds__(256) void prep_bias_kernel(const float* __restrict__ b_ih,
                                                        const float* __restrict__ b_hh,
                                                        float* __restrict__ biasArr) {
    int j = blockIdx.x * 256 + threadIdx.x;         // < 2048
    int q = j >> 6;
    int rem = j & 63;
    int g = rem >> 4;
    int m = rem & 15;
    int n = q * 16 + m;
    biasArr[j] = b_ih[g * NNODES + n] + b_hh[g * NNODES + n];
}

__global__ __launch_bounds__(256) void zero32_kernel(unsigned int* __restrict__ p, int n) {
    int i = blockIdx.x * 256 + threadIdx.x;
    if (i < n) p[i] = 0u;
}

// ---------------------------------------------------------------------------
// One LSTM timestep: G[1024,2048] = [x_t | h_in] @ WcatT^T + bias, then gate
// nonlinearities + cell update from registers. Block tile 64x256, 8 waves,
// wave tile 32x64 = 8x v_wmma_f32_16x16x32_bf16 per 32-wide K step.
// Tiles double-buffered in LDS via GLOBAL_LOAD_ASYNC_TO_LDS_B128; the
// K-loop is fully unrolled so buffers, guards, source selects and the async
// IOFFSETs are all static.
// ---------------------------------------------------------------------------
__global__ __launch_bounds__(256) void lstm_step_kernel(
    const __hip_bfloat16* __restrict__ xbf,      // [B,T,512]
    const __hip_bfloat16* __restrict__ h_in,     // [B,512]
    const __hip_bfloat16* __restrict__ wcatT,    // [2048,1024]
    const float* __restrict__ biasArr,           // [2048]
    float* __restrict__ c,                       // [B,512] in-place
    __hip_bfloat16* __restrict__ h_out,          // [B,512]
    float* __restrict__ h_f32,                   // [B,512]
    int t)
{
    __shared__ __align__(16) unsigned short ldsA[2][64][40];    // M x K tile (bf16)
    __shared__ __align__(16) unsigned short ldsB[2][256][40];   // J x K tile (bf16, B^T)

    const int tid  = threadIdx.x;
    const int lane = tid & 31;
    const int wave = tid >> 5;
    const int wm = wave >> 2;                 // 0..1 : M sub-tile (32 rows)
    const int wn = wave & 3;                  // 0..3 : J sub-tile (64 cols)
    const int b0  = blockIdx.x * 64;          // batch-row base
    const int jb0 = blockIdx.y * 256;         // gate-blocked column base

    const v8f vzero = {0.f, 0.f, 0.f, 0.f, 0.f, 0.f, 0.f, 0.f};
    v8f acc[2][4];
#pragma unroll
    for (int fm = 0; fm < 2; ++fm)
#pragma unroll
        for (int fn = 0; fn < 4; ++fn)
            acc[fm][fn] = vzero;

    // async staging: A 2048 bf16/tile -> 1 x b128/thread;
    //                B 8192 bf16/tile -> 4 x b128/thread (one row)
    const int arow  = tid >> 2;               // 0..63
    const int akseg = (tid & 3) * 8;          // 0,8,16,24 (bf16 units)
    const __hip_bfloat16* xrow =
        xbf + ((size_t)(b0 + arow) * TSTEPS + t) * NNODES + akseg;
    const __hip_bfloat16* hrow = h_in + (size_t)(b0 + arow) * NNODES + akseg;
    const __hip_bfloat16* brow = wcatT + (size_t)(jb0 + tid) * KTOT;

    const unsigned ldsAoff0 = (unsigned)(size_t)&ldsA[0][arow][akseg];
    const unsigned ldsAoff1 = (unsigned)(size_t)&ldsA[1][arow][akseg];
    const unsigned ldsBoff0 = (unsigned)(size_t)&ldsB[0][tid][0];
    const unsigned ldsBoff1 = (unsigned)(size_t)&ldsB[1][tid][0];

    // per-wave fragment indexing (ISA 16-bit A 16x32 / B 32x16 layouts)
    const int mrow  = wm * 32 + (lane & 15);
    const int kqA   = (lane >> 4) * 8;
    const int bcol0 = wn * 64 + (lane & 15);
    const int kqB   = (lane >> 4) * 8;

    // All global offsets are compile-time: tile kt is 64 bytes further along
    // its row than tile kt-1 (32 bf16); A switches base at kt==16.
#define ISSUE_TILE(KT, BUF)                                                     \
    do {                                                                        \
        const unsigned la = (BUF) ? ldsAoff1 : ldsAoff0;                        \
        const unsigned lb = (BUF) ? ldsBoff1 : ldsBoff0;                        \
        if ((KT) < 16) async_b128_off(la, xrow, (KT) * 64);                     \
        else           async_b128_off(la, hrow, ((KT) - 16) * 64);              \
        async_b128_off(lb,      brow, (KT) * 64 + 0);                           \
        async_b128_off(lb + 16, brow, (KT) * 64 + 16);                          \
        async_b128_off(lb + 32, brow, (KT) * 64 + 32);                          \
        async_b128_off(lb + 48, brow, (KT) * 64 + 48);                          \
    } while (0)

    ISSUE_TILE(0, 0);

#pragma unroll
    for (int kt = 0; kt < 32; ++kt) {
        const int cur = kt & 1;
        if (kt < 31) {
            ISSUE_TILE(kt + 1, cur ^ 1);
            // all tile-kt copies complete; only 5 prefetch copies outstanding
            asm volatile("s_wait_asynccnt 0x5" ::: "memory");
        } else {
            asm volatile("s_wait_asynccnt 0x0" ::: "memory");
        }
        __syncthreads();

        FragAB aF[2], bF[4];
        aF[0].q[0] = *reinterpret_cast<const uint4*>(&ldsA[cur][mrow][kqA]);
        aF[0].q[1] = *reinterpret_cast<const uint4*>(&ldsA[cur][mrow][16 + kqA]);
        aF[1].q[0] = *reinterpret_cast<const uint4*>(&ldsA[cur][mrow + 16][kqA]);
        aF[1].q[1] = *reinterpret_cast<const uint4*>(&ldsA[cur][mrow + 16][16 + kqA]);
#pragma unroll
        for (int fn = 0; fn < 4; ++fn) {
            bF[fn].q[0] = *reinterpret_cast<const uint4*>(&ldsB[cur][bcol0 + fn * 16][kqB]);
            bF[fn].q[1] = *reinterpret_cast<const uint4*>(&ldsB[cur][bcol0 + fn * 16][kqB + 16]);
        }

#pragma unroll
        for (int fn = 0; fn < 4; ++fn) {
            acc[0][fn] = __builtin_amdgcn_wmma_f32_16x16x32_bf16(
                false, aF[0].v, false, bF[fn].v, (short)0, acc[0][fn], false, false);
            acc[1][fn] = __builtin_amdgcn_wmma_f32_16x16x32_bf16(
                false, aF[1].v, false, bF[fn].v, (short)0, acc[1][fn], false, false);
        }
        __syncthreads();
    }
#undef ISSUE_TILE

    // Register-only LSTM epilogue: fn = gate, same lane/slot holds all 4 gates.
    const int m16 = lane & 15;
    const int jwave = jb0 + wn * 64;
    const float bi  = biasArr[jwave + 0  + m16];
    const float bfg = biasArr[jwave + 16 + m16];
    const float bgg = biasArr[jwave + 32 + m16];
    const float bog = biasArr[jwave + 48 + m16];
    const int n = ((jb0 >> 6) + wn) * 16 + m16;      // global node index

#pragma unroll
    for (int fm = 0; fm < 2; ++fm) {
#pragma unroll
        for (int r = 0; r < 8; ++r) {
            int row = wm * 32 + fm * 16 + r + ((lane >> 4) << 3);
            size_t idx = (size_t)(b0 + row) * NNODES + n;
            float gi = acc[fm][0][r] + bi;
            float gf = acc[fm][1][r] + bfg;
            float gg = acc[fm][2][r] + bgg;
            float go = acc[fm][3][r] + bog;
            float cold = c[idx];
            float ig = 1.f / (1.f + __expf(-gi));
            float fg = 1.f / (1.f + __expf(-gf));
            float og = 1.f / (1.f + __expf(-go));
            float cn = fg * cold + ig * tanhf(gg);
            float hv = og * tanhf(cn);
            c[idx] = cn;
            h_f32[idx] = hv;
            h_out[idx] = __float2bfloat16(hv);
        }
    }
}

// ---------------------------------------------------------------------------
// Head: stop-embedding segment sum + split fc2 + log_softmax + mask.
// One block (256 threads) per batch row.
// ---------------------------------------------------------------------------
__global__ __launch_bounds__(256) void head_kernel(
    const int* __restrict__ stops,              // [B,40]
    const float* __restrict__ stops_mask,       // [B,40]
    const float* __restrict__ x_dist,           // [512]
    const float* __restrict__ x_features,       // [B,8]
    const float* __restrict__ x_markov,         // [B,512]
    const int* __restrict__ x_week,             // [B]
    const unsigned char* __restrict__ x_mask,   // [B,512] bool
    const float* __restrict__ stop_emb,         // [512,64]
    const float* __restrict__ week_emb,         // [7,64]
    const float* __restrict__ fc2_w,            // [139]
    const float* __restrict__ fc2_b,            // [1]
    const float* __restrict__ h_f32,            // [B,512]
    float* __restrict__ out)                    // [B,512]
{
    __shared__ float red[256];
    const int b = blockIdx.x;
    const int tid = threadIdx.x;

    // feat . fc2_w[:136] with feat = [week_emb | x_features | masked stop-emb sum]
    float p = 0.f;
    if (tid < EDIM) {
        p = week_emb[x_week[b] * EDIM + tid] * fc2_w[tid];
    } else if (tid < EDIM + FDIM) {
        p = x_features[b * FDIM + (tid - EDIM)] * fc2_w[tid];
    } else if (tid < EDIM + FDIM + SEDIM) {
        int e = tid - (EDIM + FDIM);
        float acc = 0.f;
        for (int s = 0; s < SMAX; ++s) {
            int st = stops[b * SMAX + s];
            acc += stop_emb[st * SEDIM + e] * stops_mask[b * SMAX + s];
        }
        p = acc * fc2_w[tid];
    }
    red[tid] = p;
    __syncthreads();
    for (int s2 = 128; s2 > 0; s2 >>= 1) {
        if (tid < s2) red[tid] += red[tid + s2];
        __syncthreads();
    }
    const float featdot = red[0];
    __syncthreads();

    const float w_out = fc2_w[136], w_dist = fc2_w[137], w_mk = fc2_w[138];
    const float bb = fc2_b[0];
    const int n0 = tid, n1 = tid + 256;
    const size_t base = (size_t)b * NNODES;
    float l0 = featdot + h_f32[base + n0] * w_out + x_dist[n0] * w_dist
             + x_markov[base + n0] * w_mk + bb;
    float l1 = featdot + h_f32[base + n1] * w_out + x_dist[n1] * w_dist
             + x_markov[base + n1] * w_mk + bb;

    // log-softmax over 512 nodes
    red[tid] = fmaxf(l0, l1);
    __syncthreads();
    for (int s2 = 128; s2 > 0; s2 >>= 1) {
        if (tid < s2) red[tid] = fmaxf(red[tid], red[tid + s2]);
        __syncthreads();
    }
    const float mx = red[0];
    __syncthreads();
    red[tid] = __expf(l0 - mx) + __expf(l1 - mx);
    __syncthreads();
    for (int s2 = 128; s2 > 0; s2 >>= 1) {
        if (tid < s2) red[tid] += red[tid + s2];
        __syncthreads();
    }
    const float logZ = mx + __logf(red[0]);

    out[base + n0] = x_mask[base + n0] ? -1.0e8f : (l0 - logZ);
    out[base + n1] = x_mask[base + n1] ? -1.0e8f : (l1 - logZ);
}

// ---------------------------------------------------------------------------
// Launch
// ---------------------------------------------------------------------------
extern "C" void kernel_launch(void* const* d_in, const int* in_sizes, int n_in,
                              void* d_out, int out_size, void* d_ws, size_t ws_size,
                              hipStream_t stream) {
    (void)in_sizes; (void)n_in; (void)out_size; (void)ws_size;

    const int*   stops      = (const int*)d_in[0];
    const float* stops_mask = (const float*)d_in[1];
    const float* x          = (const float*)d_in[2];
    const float* x_dist     = (const float*)d_in[3];
    const float* x_features = (const float*)d_in[4];
    const float* x_markov   = (const float*)d_in[5];
    const int*   x_week     = (const int*)d_in[6];
    const unsigned char* x_mask = (const unsigned char*)d_in[7];
    const float* stop_emb   = (const float*)d_in[8];
    const float* week_emb   = (const float*)d_in[9];
    const float* W_ih       = (const float*)d_in[10];
    const float* W_hh       = (const float*)d_in[11];
    const float* b_ih       = (const float*)d_in[12];
    const float* b_hh       = (const float*)d_in[13];
    const float* fc2_w      = (const float*)d_in[14];
    const float* fc2_b      = (const float*)d_in[15];
    float* out = (float*)d_out;

    char* ws = (char*)d_ws;
    __hip_bfloat16* xbf   = (__hip_bfloat16*)(ws + XBF_OFF);
    __hip_bfloat16* wcatT = (__hip_bfloat16*)(ws + WCAT_OFF);
    float*          bias  = (float*)(ws + BIAS_OFF);
    __hip_bfloat16* hA    = (__hip_bfloat16*)(ws + HA_OFF);
    __hip_bfloat16* hB    = (__hip_bfloat16*)(ws + HB_OFF);
    float*          cbuf  = (float*)(ws + C_OFF);
    float*          hf32  = (float*)(ws + HF_OFF);

    // --- prep ---
    {
        int n4 = (BATCH * TSTEPS * NNODES) / 4;                 // 7,864,320
        prep_x_kernel<<<(n4 + 255) / 256, 256, 0, stream>>>(x, xbf, n4);
    }
    prep_w_kernel<<<(JTOT * KTOT) / 256, 256, 0, stream>>>(W_ih, W_hh, wcatT);
    prep_bias_kernel<<<JTOT / 256, 256, 0, stream>>>(b_ih, b_hh, bias);
    zero32_kernel<<<(BATCH * NNODES) / 256, 256, 0, stream>>>(
        (unsigned int*)cbuf, BATCH * NNODES);                   // c = 0
    zero32_kernel<<<(BATCH * NNODES / 2) / 256, 256, 0, stream>>>(
        (unsigned int*)hA, BATCH * NNODES / 2);                 // h0 = 0 (bf16)

    // --- 60 sequential LSTM steps, h double-buffered ---
    dim3 grid(BATCH / 64, JTOT / 256);                          // 16 x 8
    for (int t = 0; t < TSTEPS; ++t) {
        const __hip_bfloat16* hin = (t & 1) ? hB : hA;
        __hip_bfloat16*       hout = (t & 1) ? hA : hB;
        lstm_step_kernel<<<grid, 256, 0, stream>>>(
            xbf, hin, wcatT, bias, cbuf, hout, hf32, t);
    }

    // --- head: fc2 + log_softmax + mask ---
    head_kernel<<<BATCH, 256, 0, stream>>>(
        stops, stops_mask, x_dist, x_features, x_markov, x_week, x_mask,
        stop_emb, week_emb, fc2_w, fc2_b, hf32, out);
}
